// VarianceAdaptor_55783035240931
// MI455X (gfx1250) — matrix operand order repaired and checked by reading, not compiled, not run on hardware
//
#include <hip/hip_runtime.h>
#include <hip/hip_bf16.h>

typedef __attribute__((ext_vector_type(16))) _Float16 v16h;
typedef __attribute__((ext_vector_type(8)))  float    v8f;
typedef __attribute__((ext_vector_type(4)))  unsigned int u32x4;
typedef __attribute__((ext_vector_type(8)))  unsigned int u32x8;

#define NB 48
#define NT 512
#define NH 256
#define NL 2048
#define NROW (NB*NT)          // 24576
#define PE_ROWS (NL+1)        // 2049
#define LN_EPS 1e-5f

// ---------------------------------------------------------------------------
// Weight packing: conv weight w[o][i][k] (f32) -> WMMA-B-fragment order f16:
// out[((k*8 + c)*16 + ot)*32*16 + lane*16 + e]
//   o  = ot*16 + (lane&15)
//   ci = c*32 + 16*(lane>>4) + e      (contiguous k order, same as the A fragment reads)
// ---------------------------------------------------------------------------
__global__ void pack_conv_w_kernel(const float* __restrict__ w,
                                   _Float16* __restrict__ out, int nk) {
  int idx = blockIdx.x * blockDim.x + threadIdx.x;
  int total = nk * 8 * 16 * 32 * 16;
  if (idx >= total) return;
  int e    = idx & 15;
  int lane = (idx >> 4) & 31;
  int ot   = (idx >> 9) & 15;
  int c    = (idx >> 13) & 7;
  int k    = idx >> 16;
  int o  = ot * 16 + (lane & 15);
  int ci = c * 32 + 16 * (lane >> 4) + e;
  out[idx] = (_Float16)w[((size_t)o * NH + ci) * nk + k];
}

// fw[i][o] (f32, i-major) -> same fragment order (k==0 only)
__global__ void pack_fw_kernel(const float* __restrict__ w,
                               _Float16* __restrict__ out) {
  int idx = blockIdx.x * blockDim.x + threadIdx.x;
  if (idx >= 8 * 16 * 32 * 16) return;
  int e    = idx & 15;
  int lane = (idx >> 4) & 31;
  int ot   = (idx >> 9) & 15;
  int c    = (idx >> 13) & 7;
  int o  = ot * 16 + (lane & 15);
  int ci = c * 32 + 16 * (lane >> 4) + e;
  out[idx] = (_Float16)w[(size_t)ci * NH + o];
}

__global__ void cvt_f16_kernel(const float* __restrict__ in,
                               _Float16* __restrict__ out, int n) {
  int i = blockIdx.x * blockDim.x + threadIdx.x;
  if (i < n) out[i] = (_Float16)in[i];
}

__global__ void relu_f16_kernel(const float* __restrict__ in,
                                _Float16* __restrict__ out, int n) {
  int i = blockIdx.x * blockDim.x + threadIdx.x;
  if (i < n) out[i] = (_Float16)fmaxf(in[i], 0.0f);
}

// ---------------------------------------------------------------------------
// WMMA conv1d (NK taps, 'same' padding) / GEMM (NK==1).
// Block = 256 threads = 8 waves; block computes a 16-row x 256-col slab.
// X slab (16+NK-1 rows x 256 ch, f16) staged ONCE per block by the Tensor
// Data Mover into LDS.  TDM LDS padding (2 DWORDs per 128-DWORD row) gives a
// 520-byte row stride so the 16-lane A-fragment reads are bank-decorrelated.
// Wave w computes output tiles (16x16) at columns [32w, 32w+32).
// Y = X * Wpacked + bias  (f32 out)
// ---------------------------------------------------------------------------
template<int NK>
__global__ __launch_bounds__(256) void conv_wmma_kernel(
    const _Float16* __restrict__ X, const _Float16* __restrict__ Wp,
    const float* __restrict__ bias, float* __restrict__ Y) {
  constexpr int NR = 16 + NK - 1;     // rows incl. halo
  __shared__ _Float16 ldsX[NR][260];  // 520B stride == TDM pad layout
  const int tile = blockIdx.x;
  const int row0 = tile * 16;         // 512 % 16 == 0 -> tile stays in one b
  const int bb   = row0 / NT;
  const int t0   = row0 % NT;
  const int tid  = threadIdx.x;
  const int wave = tid >> 5;
  const int lane = tid & 31;
  const int m    = lane & 15;
  const int half = lane >> 4;
  const int ot0  = wave * 2;
  const int ot1  = ot0 + 1;

  // ---- clamp halo at batch edges; zero rows the TDM won't write ----
  int rlo = 0, tstart = t0, nrows = NR;
  if (NK == 3) {
    rlo    = (t0 == 0) ? 1 : 0;
    tstart = t0 - 1 + rlo;
    nrows  = NR - rlo - ((t0 == NT - 16) ? 1 : 0);
    if (t0 == 0)       ldsX[0][tid]      = (_Float16)0.0f;   // tid==ch, 256 wide
    if (t0 == NT - 16) ldsX[NR - 1][tid] = (_Float16)0.0f;
  }

  // ---- one 2D TDM descriptor per block, issued by wave 0 ----
  if (tid < 32) {
    const unsigned lbase =
        (unsigned)(size_t)(&ldsX[0][0]) + (unsigned)(rlo * 260 * 2);
    const unsigned long long ga =
        (unsigned long long)(size_t)(X + ((size_t)bb * NT + tstart) * NH);
    u32x4 g0;
    g0[0] = 1u;                                  // count=1, user descriptor
    g0[1] = lbase;                               // lds_addr (bytes)
    g0[2] = (unsigned)(ga & 0xffffffffu);        // global_addr[31:0]
    g0[3] = (unsigned)(ga >> 32) | (2u << 30);   // global_addr[56:32] | type=2
    u32x8 g1;
    g1[0] = (1u << 16)      // data_size = 2 bytes
          | (1u << 20)      // pad_enable
          | (6u << 22)      // pad_interval: 128 DWORDs (one 512B row)
          | (1u << 25);     // pad_amount: 2 DWORDs (row stride -> 520B)
    g1[1] = ((unsigned)NH) << 16;        // tensor_dim0[15:0] = 256
    g1[2] = ((unsigned)nrows) << 16;     // tensor_dim0 hi=0 | tensor_dim1 lo
    g1[3] = ((unsigned)NH) << 16;        // tensor_dim1 hi=0 | tile_dim0 = 256
    g1[4] = (unsigned)nrows;             // tile_dim1 | tile_dim2 = 0
    g1[5] = (unsigned)NH;                // tensor_dim0_stride = 256 elements
    g1[6] = 0u;                          // stride0 hi | tensor_dim1_stride lo
    g1[7] = 0u;
    asm volatile("tensor_load_to_lds %0, %1" :: "s"(g0), "s"(g1) : "memory");
    __builtin_amdgcn_s_wait_tensorcnt(0);
  }
  __syncthreads();

  v8f acc0 = {0.f, 0.f, 0.f, 0.f, 0.f, 0.f, 0.f, 0.f};
  v8f acc1 = acc0;

  for (int c = 0; c < 8; ++c) {
#pragma unroll
    for (int k = 0; k < NK; ++k) {
      // A fragment: element e -> K = 16*half + e (contiguous, matches packed B);
      // row M = lane&15, shifted by tap k.
      v16h a;
      const _Float16* p0 = &ldsX[m + k][c * 32 + 16 * half];
      const _Float16* p1 = &ldsX[m + k][c * 32 + 16 * half + 8];
#pragma unroll
      for (int e = 0; e < 8; ++e) { a[e] = p0[e]; a[8 + e] = p1[e]; }
      // B fragments: pre-swizzled -> one contiguous 32B load per tile
      const size_t kc = (size_t)(k * 8 + c);
      const v16h b0 = *(const v16h*)(Wp + ((kc * 16 + ot0) * 32 + lane) * 16);
      const v16h b1 = *(const v16h*)(Wp + ((kc * 16 + ot1) * 32 + lane) * 16);
      acc0 = __builtin_amdgcn_wmma_f32_16x16x32_f16(false, a, false, b0,
                                                    (short)0, acc0, false, false);
      acc1 = __builtin_amdgcn_wmma_f32_16x16x32_f16(false, a, false, b1,
                                                    (short)0, acc1, false, false);
    }
  }

  // C/D layout: VGPR r -> M = r + 8*half, N = lane&15
  const int n = lane & 15;
#pragma unroll
  for (int r = 0; r < 8; ++r) {
    const int M = r + 8 * half;
    const size_t orow = (size_t)(row0 + M) * NH;
    const int o0 = ot0 * 16 + n;
    const int o1 = ot1 * 16 + n;
    Y[orow + o0] = acc0[r] + bias[o0];
    Y[orow + o1] = acc1[r] + bias[o1];
  }
}

// ---------------------------------------------------------------------------
// LayerNorm(ReLU(x)) over H=256; optional f32 and/or f16 outputs.
// ---------------------------------------------------------------------------
__global__ __launch_bounds__(256) void ln_relu_kernel(
    const float* __restrict__ in, const float* __restrict__ g,
    const float* __restrict__ be, float* __restrict__ o32,
    _Float16* __restrict__ o16) {
  const int row = blockIdx.x;
  const int o   = threadIdx.x;
  float v = fmaxf(in[(size_t)row * NH + o], 0.0f);
  __shared__ float red[16];
  __shared__ float mv[2];
  float s = v, q = v * v;
#pragma unroll
  for (int off = 16; off > 0; off >>= 1) {
    s += __shfl_xor(s, off, 32);
    q += __shfl_xor(q, off, 32);
  }
  const int wave = threadIdx.x >> 5, lane = threadIdx.x & 31;
  if (lane == 0) { red[wave] = s; red[8 + wave] = q; }
  __syncthreads();
  if (threadIdx.x == 0) {
    float S = 0.f, Q = 0.f;
    for (int i = 0; i < 8; ++i) { S += red[i]; Q += red[8 + i]; }
    float mean = S * (1.0f / NH);
    mv[0] = mean;
    mv[1] = Q * (1.0f / NH) - mean * mean;
  }
  __syncthreads();
  float y = (v - mv[0]) * rsqrtf(mv[1] + LN_EPS) * g[o] + be[o];
  if (o32) o32[(size_t)row * NH + o] = y;
  if (o16) o16[(size_t)row * NH + o] = (_Float16)y;
}

// out[row] = mask ? 0 : dot(act[row], lw) + lb
__global__ __launch_bounds__(256) void dot_mask_kernel(
    const float* __restrict__ act, const float* __restrict__ lw,
    const float* __restrict__ lb, const unsigned char* __restrict__ mask,
    float* __restrict__ out) {
  const int row = blockIdx.x;
  const int o   = threadIdx.x;
  float v = act[(size_t)row * NH + o] * lw[o];
  __shared__ float red[8];
#pragma unroll
  for (int off = 16; off > 0; off >>= 1) v += __shfl_xor(v, off, 32);
  const int wave = threadIdx.x >> 5, lane = threadIdx.x & 31;
  if (lane == 0) red[wave] = v;
  __syncthreads();
  if (threadIdx.x == 0) {
    float S = 0.f;
    for (int i = 0; i < 8; ++i) S += red[i];
    out[row] = mask[row] ? 0.0f : (S + lb[0]);
  }
}

// variance_embedding first conv: scalar input, k=3, ReLU -> f16
__global__ __launch_bounds__(256) void emb_conv1_kernel(
    const float* __restrict__ v, const float* __restrict__ w1,
    const float* __restrict__ b1, _Float16* __restrict__ out) {
  const int row = blockIdx.x;
  const int o   = threadIdx.x;
  const int t   = row % NT;
  float vc = v[row];
  float vp = (t > 0)      ? v[row - 1] : 0.0f;
  float vn = (t < NT - 1) ? v[row + 1] : 0.0f;
  float y = w1[o * 3 + 0] * vp + w1[o * 3 + 1] * vc + w1[o * 3 + 2] * vn + b1[o];
  out[(size_t)row * NH + o] = (_Float16)fmaxf(y, 0.0f);
}

// h = LayerNorm(x; g, b) + pitch_emb + energy_emb
__global__ __launch_bounds__(256) void combine_kernel(
    const float* __restrict__ x, const float* __restrict__ g,
    const float* __restrict__ be, const float* __restrict__ pemb,
    const float* __restrict__ eemb, float* __restrict__ h) {
  const int row = blockIdx.x;
  const int o   = threadIdx.x;
  const size_t i = (size_t)row * NH + o;
  float v = x[i];
  __shared__ float red[16];
  __shared__ float mv[2];
  float s = v, q = v * v;
#pragma unroll
  for (int off = 16; off > 0; off >>= 1) {
    s += __shfl_xor(s, off, 32);
    q += __shfl_xor(q, off, 32);
  }
  const int wave = threadIdx.x >> 5, lane = threadIdx.x & 31;
  if (lane == 0) { red[wave] = s; red[8 + wave] = q; }
  __syncthreads();
  if (threadIdx.x == 0) {
    float S = 0.f, Q = 0.f;
    for (int i2 = 0; i2 < 8; ++i2) { S += red[i2]; Q += red[8 + i2]; }
    float mean = S * (1.0f / NH);
    mv[0] = mean;
    mv[1] = Q * (1.0f / NH) - mean * mean;
  }
  __syncthreads();
  h[i] = (v - mv[0]) * rsqrtf(mv[1] + LN_EPS) * g[o] + be[o] + pemb[i] + eemb[i];
}

// sinusoid PE table: block = pos (2049), thread = d (256)
__global__ __launch_bounds__(256) void pe_table_kernel(float* __restrict__ pet) {
  const int pos = blockIdx.x;
  const int d   = threadIdx.x;
  float expo  = (float)(2 * (d / 2)) / (float)NH;
  float angle = (float)pos * powf(10000.0f, -expo);
  pet[(size_t)pos * NH + d] = (d & 1) ? cosf(angle) : sinf(angle);
}

// inclusive cumsum of duration per batch row; also emits mel_len (as float)
__global__ __launch_bounds__(512) void cumsum_kernel(
    const int* __restrict__ dur, int* __restrict__ cum,
    float* __restrict__ mel_len) {
  __shared__ int s[NT];
  const int b = blockIdx.x;
  const int t = threadIdx.x;
  s[t] = dur[(size_t)b * NT + t];
  __syncthreads();
  for (int off = 1; off < NT; off <<= 1) {
    int v = (t >= off) ? s[t - off] : 0;
    __syncthreads();
    s[t] += v;
    __syncthreads();
  }
  cum[(size_t)b * NT + t] = s[t];
  if (t == NT - 1) mel_len[b] = (float)s[NT - 1];
}

// length regulator: block = (b, t_out), thread = channel
__global__ __launch_bounds__(256) void regulate_kernel(
    const float* __restrict__ h, const int* __restrict__ cum,
    const int* __restrict__ dur, const float* __restrict__ pet,
    float* __restrict__ out, float* __restrict__ melmask) {
  const int blk = blockIdx.x;
  const int b = blk / NL;
  const int t = blk % NL;
  const int* cb = cum + (size_t)b * NT;
  int lo = 0, hi = NT;                       // searchsorted(cum, t, 'right')
  while (lo < hi) {
    int mid = (lo + hi) >> 1;
    if (cb[mid] <= t) lo = mid + 1; else hi = mid;
  }
  const int tokc = (lo < NT - 1) ? lo : (NT - 1);
  const int mel  = cb[NT - 1];
  const bool valid = t < mel;
  int pos = t - (cb[tokc] - dur[(size_t)b * NT + tokc]);
  pos = pos < 0 ? 0 : (pos > NL ? NL : pos);
  const int o = threadIdx.x;
  float val = 0.0f;
  if (valid)
    val = h[((size_t)b * NT + tokc) * NH + o] + pet[(size_t)pos * NH + o];
  out[((size_t)b * NL + t) * NH + o] = val;
  if (o == 0) melmask[(size_t)b * NL + t] = valid ? 0.0f : 1.0f;
}

// ---------------------------------------------------------------------------
extern "C" void kernel_launch(void* const* d_in, const int* in_sizes, int n_in,
                              void* d_out, int out_size, void* d_ws, size_t ws_size,
                              hipStream_t stream) {
  (void)in_sizes; (void)n_in; (void)out_size; (void)ws_size;

  const float*         x    = (const float*)d_in[0];
  const unsigned char* mask = (const unsigned char*)d_in[1];
  const int*           dur  = (const int*)d_in[2];
  // d_in[3] = max_len scalar (constant 2048, unused)
  static const int PRED_BASE[3] = {4, 14, 24};   // dur_p, pitch_p, energy_p
  static const int EMB_BASE[2]  = {34, 40};      // pitch_e, energy_e
  const float* ln_g = (const float*)d_in[46];
  const float* ln_b = (const float*)d_in[47];

  // ---- workspace carve ----
  char* ws = (char*)d_ws;
  size_t off = 0;
  auto carve = [&](size_t bytes) -> void* {
    void* p = ws + off;
    off += (bytes + 255) & ~(size_t)255;
    return p;
  };
  _Float16* xf16  = (_Float16*)carve((size_t)NROW * NH * 2);
  _Float16* wp[8];
  for (int i = 0; i < 8; ++i) wp[i] = (_Float16*)carve((size_t)3 * 65536 * 2);
  _Float16* fwp[2];
  for (int i = 0; i < 2; ++i) fwp[i] = (_Float16*)carve((size_t)65536 * 2);
  float*    bufA  = (float*)   carve((size_t)NROW * NH * 4);
  _Float16* actH  = (_Float16*)carve((size_t)NROW * NH * 2);
  float*    act2  = (float*)   carve((size_t)NROW * NH * 4);
  float*    pemb  = (float*)   carve((size_t)NROW * NH * 4);
  float*    eemb  = (float*)   carve((size_t)NROW * NH * 4);
  float*    hbuf  = (float*)   carve((size_t)NROW * NH * 4);
  float*    pet   = (float*)   carve((size_t)PE_ROWS * NH * 4);
  int*      cum   = (int*)     carve((size_t)NB * NT * 4);

  // ---- output layout (flat, return order) ----
  float* out_main    = (float*)d_out;                       // (B, L, H)
  float* out_logdur  = out_main + (size_t)NB * NL * NH;     // (B, T)
  float* out_pitch   = out_logdur + (size_t)NB * NT;
  float* out_energy  = out_pitch + (size_t)NB * NT;
  float* out_mellen  = out_energy + (size_t)NB * NT;        // (B,)
  float* out_melmask = out_mellen + NB;                     // (B, L)

  // ---- weight packing (amortized; weights reused over 24576 rows) ----
  for (int p = 0; p < 3; ++p) {
    pack_conv_w_kernel<<<(3 * 65536 + 255) / 256, 256, 0, stream>>>(
        (const float*)d_in[PRED_BASE[p] + 0], wp[2 * p + 0], 3);
    pack_conv_w_kernel<<<(3 * 65536 + 255) / 256, 256, 0, stream>>>(
        (const float*)d_in[PRED_BASE[p] + 4], wp[2 * p + 1], 3);
  }
  for (int e = 0; e < 2; ++e) {
    pack_conv_w_kernel<<<(3 * 65536 + 255) / 256, 256, 0, stream>>>(
        (const float*)d_in[EMB_BASE[e] + 2], wp[6 + e], 3);
    pack_fw_kernel<<<65536 / 256, 256, 0, stream>>>(
        (const float*)d_in[EMB_BASE[e] + 4], fwp[e]);
  }

  // ---- precompute ----
  const int nelem = NROW * NH;
  cvt_f16_kernel<<<(nelem + 255) / 256, 256, 0, stream>>>(x, xf16, nelem);
  pe_table_kernel<<<PE_ROWS, 256, 0, stream>>>(pet);
  cumsum_kernel<<<NB, NT, 0, stream>>>(dur, cum, out_mellen);

  // ---- three variance predictors ----
  float* pouts[3] = {out_logdur, out_pitch, out_energy};
  for (int p = 0; p < 3; ++p) {
    const int base = PRED_BASE[p];
    const float* b1  = (const float*)d_in[base + 1];
    const float* g1  = (const float*)d_in[base + 2];
    const float* be1 = (const float*)d_in[base + 3];
    const float* b2  = (const float*)d_in[base + 5];
    const float* g2  = (const float*)d_in[base + 6];
    const float* be2 = (const float*)d_in[base + 7];
    const float* lw  = (const float*)d_in[base + 8];
    const float* lb  = (const float*)d_in[base + 9];

    conv_wmma_kernel<3><<<NROW / 16, 256, 0, stream>>>(xf16, wp[2 * p], b1, bufA);
    ln_relu_kernel<<<NROW, 256, 0, stream>>>(bufA, g1, be1, (float*)nullptr, actH);
    conv_wmma_kernel<3><<<NROW / 16, 256, 0, stream>>>(actH, wp[2 * p + 1], b2, bufA);
    ln_relu_kernel<<<NROW, 256, 0, stream>>>(bufA, g2, be2, act2, (_Float16*)nullptr);
    dot_mask_kernel<<<NROW, 256, 0, stream>>>(act2, lw, lb, mask, pouts[p]);
  }

  // ---- pitch / energy embeddings ----
  for (int e = 0; e < 2; ++e) {
    const int base = EMB_BASE[e];
    const float* w1 = (const float*)d_in[base + 0];
    const float* b1 = (const float*)d_in[base + 1];
    const float* b2 = (const float*)d_in[base + 3];
    const float* fb = (const float*)d_in[base + 5];
    const float* v  = (e == 0) ? out_pitch : out_energy;
    float* emb_out  = (e == 0) ? pemb : eemb;

    emb_conv1_kernel<<<NROW, 256, 0, stream>>>(v, w1, b1, actH);
    conv_wmma_kernel<3><<<NROW / 16, 256, 0, stream>>>(actH, wp[6 + e], b2, bufA);
    relu_f16_kernel<<<(nelem + 255) / 256, 256, 0, stream>>>(bufA, actH, nelem);
    conv_wmma_kernel<1><<<NROW / 16, 256, 0, stream>>>(actH, fwp[e], fb, emb_out);
  }

  // ---- h = LN(x) + pitch_emb + energy_emb; then length-regulate ----
  combine_kernel<<<NROW, 256, 0, stream>>>(x, ln_g, ln_b, pemb, eemb, hbuf);
  regulate_kernel<<<NB * NL, 256, 0, stream>>>(hbuf, cum, dur, pet,
                                               out_main, out_melmask);
}
